// Branch1_71725953843343
// MI455X (gfx1250) — compile-verified, matrix operand order checked by reading, not run on hardware
//
#include <hip/hip_runtime.h>
#include <hip/hip_bf16.h>

typedef __attribute__((ext_vector_type(16))) __bf16 v16bf;
typedef __attribute__((ext_vector_type(8)))  float  v8f;

#define NB    1024
#define C1    103
#define IN1   (C1*15*15)     // 23175
#define P1    169            // 13x13
#define K1    927
#define OC1   32
#define KB1   29             // k-blocks of 32 (927 -> 928)
#define MT1   11             // M tiles (169 -> 176)
#define NT1   2

#define K2    288            // 32*9
#define OC2   64
#define P2    121            // 11x11
#define KB2   9
#define MT2   8              // 121 -> 128
#define NT2   4

#define X6SZ  (NB*256)

// fragment element e + lane-half -> K offset within a 32-wide k-block
__device__ __forceinline__ int kidx(int e, int half) {
  return ((e < 8) ? 0 : 16) + half * 8 + (e & 7);
}

__device__ __forceinline__ v8f wmma_bf16(v16bf a, v16bf b, v8f c) {
  return __builtin_amdgcn_wmma_f32_16x16x32_bf16(
             false, a, false, b, (short)0, c, false, false);
}

// -------- weight prep: fp32 OIHW -> bf16 fragments [nt][kb][lane][16] --------
__global__ __launch_bounds__(256) void prep_weights(
    const float* __restrict__ w1, const float* __restrict__ w2,
    __bf16* __restrict__ w1f, __bf16* __restrict__ w2f)
{
  const int n1f = NT1 * KB1 * 32 * 16;   // 29696
  const int n2f = NT2 * KB2 * 32 * 16;   // 18432
  int i = blockIdx.x * blockDim.x + threadIdx.x;
  if (i < n1f) {
    int e = i & 15, ln = (i >> 4) & 31, q = i >> 9;   // q = nt*KB1+kbi
    int nt = q / KB1, kbi = q - nt * KB1;
    int o = nt * 16 + (ln & 15);
    int k = kbi * 32 + kidx(e, ln >> 4);
    w1f[i] = (k < K1) ? (__bf16)w1[o * K1 + k] : (__bf16)0.0f;
  } else if (i < n1f + n2f) {
    int j = i - n1f;
    int e = j & 15, ln = (j >> 4) & 31, q = j >> 9;
    int nt = q / KB2, kbi = q - nt * KB2;
    int o = nt * 16 + (ln & 15);
    int k = kbi * 32 + kidx(e, ln >> 4);
    w2f[j] = (__bf16)w2[o * K2 + k];
  }
}

// ---------------- conv1 implicit GEMM (WMMA bf16) ----------------
__global__ __launch_bounds__(256) void conv1_kernel(
    const float* __restrict__ x, const __bf16* __restrict__ w1f,
    const float* __restrict__ cb,
    const float* __restrict__ g, const float* __restrict__ bb,
    const float* __restrict__ bm, const float* __restrict__ bv,
    __bf16* __restrict__ x1bf)
{
  __shared__ __attribute__((aligned(32))) __bf16 xs[IN1];             // 46350 B
  __shared__ __attribute__((aligned(32))) __bf16 panel[MT1*32*16];    // 11264 B
  __shared__ unsigned short offT[K1];                                 // 1854 B

  const int b = blockIdx.x;
  const int tid = threadIdx.x;
  const float* xb = x + (size_t)b * IN1;
  for (int i = tid; i < IN1; i += 256) xs[i] = (__bf16)xb[i];
  for (int k = tid; k < K1; k += 256) {
    int c = k / 9, t = k - c * 9;
    offT[k] = (unsigned short)(c * 225 + (t / 3) * 15 + (t % 3));
  }
  __syncthreads();

  const int wave = tid >> 5;
  const int lane = tid & 31;
  const int l16  = lane & 15;
  const int half = lane >> 4;

  // ---- panel-build thread constants (slot i = tid + j*256) ----
  const int e     = tid & 15;        // fragment element (low 4 bits)
  const int r16   = tid >> 4;        // 0..15 -> row within M tile
  const int koff0 = kidx(e, 0);
  const int koff1 = koff0 + 8;       // kidx(e,1)
  int  mbv[MT1];
  bool mok[MT1];
#pragma unroll
  for (int jh = 0; jh < MT1; ++jh) {
    const int m = jh * 16 + r16;
    mok[jh] = (m < P1);
    mbv[jh] = mok[jh] ? ((m / 13) * 15 + (m % 13)) : 0;
  }

  // per-wave tiling: nt fixed per wave, three M tiles (last invalid for waves 6,7)
  const int nt  = wave & 1;
  const int mt0 = wave >> 1;                // + {0,4,8}
  const bool t2ok = (wave < 6);
  const int mt2 = t2ok ? (mt0 + 8) : 0;     // dummy tile -> discarded

  const __bf16* bptr = w1f + ((size_t)(nt * KB1) * 32 + lane) * 16;

  v8f acc0 = {}, acc1 = {}, acc2 = {};

  for (int kbi = 0; kbi < KB1; ++kbi) {
    // 2 offT lookups shared by all M tiles; branchless build
    const int k0 = kbi * 32 + koff0;
    const int k1 = kbi * 32 + koff1;
    const bool k0ok = (k0 < K1), k1ok = (k1 < K1);
    const int o0 = offT[k0ok ? k0 : 0];
    const int o1 = offT[k1ok ? k1 : 0];
#pragma unroll
    for (int jh = 0; jh < MT1; ++jh) {
      __bf16 v0 = xs[o0 + mbv[jh]];
      __bf16 v1 = xs[o1 + mbv[jh]];
      panel[tid + (2 * jh) * 256]     = (mok[jh] && k0ok) ? v0 : (__bf16)0.0f;
      panel[tid + (2 * jh + 1) * 256] = (mok[jh] && k1ok) ? v1 : (__bf16)0.0f;
    }
    __syncthreads();

    // one B fragment per wave, three A fragments, three back-to-back WMMAs
    v16bf Bv = *(const v16bf*)(bptr + (size_t)kbi * (32 * 16));
    v16bf A0 = *(const v16bf*)(panel + ((mt0 + 0) * 32 + lane) * 16);
    v16bf A1 = *(const v16bf*)(panel + ((mt0 + 4) * 32 + lane) * 16);
    v16bf A2 = *(const v16bf*)(panel + (mt2 * 32 + lane) * 16);
    acc0 = wmma_bf16(A0, Bv, acc0);
    acc1 = wmma_bf16(A1, Bv, acc1);
    acc2 = wmma_bf16(A2, Bv, acc2);
    __syncthreads();   // panel reused next iteration
  }

  // fused epilogue: +bias, leaky(0.01), BN1, store bf16
  const int n = nt * 16 + l16;
  const float scale = g[n] * rsqrtf(bv[n] + 1e-5f);
  const float shift = bb[n] - bm[n] * scale;
  const float bias  = cb[n];
#pragma unroll
  for (int ti = 0; ti < 3; ++ti) {
    if (ti == 2 && !t2ok) break;
    const int mt = (ti == 2) ? mt2 : (mt0 + ti * 4);
    const v8f acc = (ti == 0) ? acc0 : (ti == 1) ? acc1 : acc2;
#pragma unroll
    for (int r = 0; r < 8; ++r) {
      const int mm = mt * 16 + r + half * 8;
      if (mm < P1) {
        float v = acc[r] + bias;
        v = (v >= 0.f) ? v : 0.01f * v;
        v = v * scale + shift;
        x1bf[((size_t)b * OC1 + n) * P1 + mm] = (__bf16)v;
      }
    }
  }
}

// ---------------- conv2 (WMMA) + fused graph head ----------------
__global__ __launch_bounds__(256) void conv2_graph_kernel(
    const __bf16* __restrict__ x1bf, const __bf16* __restrict__ w2f,
    const float* __restrict__ cb,
    const float* __restrict__ g, const float* __restrict__ bb,
    const float* __restrict__ bm, const float* __restrict__ bv,
    const float* __restrict__ fc_sam,
    const float* __restrict__ ctw, const float* __restrict__ ctb,
    const float* __restrict__ adjw,
    const float* __restrict__ sww, const float* __restrict__ swb,
    const float* __restrict__ dcow, const float* __restrict__ dcob,
    const float* __restrict__ dgam,
    const float* __restrict__ dww, const float* __restrict__ dwb,
    float* __restrict__ x6out, float* __restrict__ x2out)
{
  __shared__ __attribute__((aligned(32))) __bf16 x1s[OC1 * P1];       // 10816
  __shared__ __attribute__((aligned(32))) __bf16 panel[MT2*32*16];    // 8192
  __shared__ unsigned short offT[K2];                                 // 576
  __shared__ float  x2s[OC2 * P2];     // 30976
  __shared__ float  masks[4 * P2];     // 1936
  __shared__ float  Gm[OC2 * 4];       // reused as 'a'
  __shared__ float  x3s[OC2 * 4];
  __shared__ float  x4s[OC2 * 4];
  __shared__ float  outs[OC2 * 4];
  __shared__ float  y1s[OC2 * 4];
  __shared__ float  dadj[16];
  __shared__ float  sn[4];

  const int b = blockIdx.x;
  const int tid = threadIdx.x;
  for (int i = tid; i < OC1 * P1; i += 256)
    x1s[i] = x1bf[(size_t)b * OC1 * P1 + i];
  for (int k = tid; k < K2; k += 256) {
    int c = k / 9, t = k - c * 9;
    offT[k] = (unsigned short)(c * P1 + (t / 3) * 13 + (t % 3));
  }
  __syncthreads();

  const int wave = tid >> 5;
  const int lane = tid & 31;
  const int l16  = lane & 15;
  const int half = lane >> 4;

  // panel-build thread constants
  const int e     = tid & 15;
  const int r16   = tid >> 4;
  const int koff0 = kidx(e, 0);
  const int koff1 = koff0 + 8;
  int  mbv[MT2];
  bool mok[MT2];
#pragma unroll
  for (int jh = 0; jh < MT2; ++jh) {
    const int m = jh * 16 + r16;
    mok[jh] = (m < P2);
    mbv[jh] = mok[jh] ? ((m / 11) * 13 + (m % 11)) : 0;
  }

  // nt fixed per wave; M tiles mt0 + {0,2,4,6} (all valid)
  const int nt  = wave & 3;
  const int mt0 = wave >> 2;
  const __bf16* bptr = w2f + ((size_t)(nt * KB2) * 32 + lane) * 16;

  v8f acc0 = {}, acc1 = {}, acc2 = {}, acc3 = {};

  for (int kbi = 0; kbi < KB2; ++kbi) {
    const int o0 = offT[kbi * 32 + koff0];
    const int o1 = offT[kbi * 32 + koff1];
#pragma unroll
    for (int jh = 0; jh < MT2; ++jh) {
      __bf16 v0 = x1s[o0 + mbv[jh]];
      __bf16 v1 = x1s[o1 + mbv[jh]];
      panel[tid + (2 * jh) * 256]     = mok[jh] ? v0 : (__bf16)0.0f;
      panel[tid + (2 * jh + 1) * 256] = mok[jh] ? v1 : (__bf16)0.0f;
    }
    __syncthreads();

    v16bf Bv = *(const v16bf*)(bptr + (size_t)kbi * (32 * 16));
    v16bf A0 = *(const v16bf*)(panel + ((mt0 + 0) * 32 + lane) * 16);
    v16bf A1 = *(const v16bf*)(panel + ((mt0 + 2) * 32 + lane) * 16);
    v16bf A2 = *(const v16bf*)(panel + ((mt0 + 4) * 32 + lane) * 16);
    v16bf A3 = *(const v16bf*)(panel + ((mt0 + 6) * 32 + lane) * 16);
    acc0 = wmma_bf16(A0, Bv, acc0);
    acc1 = wmma_bf16(A1, Bv, acc1);
    acc2 = wmma_bf16(A2, Bv, acc2);
    acc3 = wmma_bf16(A3, Bv, acc3);
    __syncthreads();
  }

  // epilogue: +bias, leaky(0.01), BN2 -> LDS + d_out
  {
    const int n = nt * 16 + l16;
    const float scale = g[n] * rsqrtf(bv[n] + 1e-5f);
    const float shift = bb[n] - bm[n] * scale;
    const float bias  = cb[n];
#pragma unroll
    for (int ti = 0; ti < 4; ++ti) {
      const int mt = mt0 + ti * 2;
      const v8f acc = (ti == 0) ? acc0 : (ti == 1) ? acc1 : (ti == 2) ? acc2 : acc3;
#pragma unroll
      for (int r = 0; r < 8; ++r) {
        const int mm = mt * 16 + r + half * 8;
        if (mm < P2) {
          float v = acc[r] + bias;
          v = (v >= 0.f) ? v : 0.01f * v;
          v = v * scale + shift;
          x2s[n * P2 + mm] = v;
          x2out[(size_t)b * OC2 * P2 + n * P2 + mm] = v;
        }
      }
    }
  }
  __syncthreads();

  // mask[n][l] = sigmoid(fc_sam[n]·flat[:,l])
  if (tid < P2) {
    const int l = tid;
#pragma unroll
    for (int n = 0; n < 4; ++n) {
      float s = 0.f;
      for (int c = 0; c < OC2; ++c) s += fc_sam[n * OC2 + c] * x2s[c * P2 + l];
      masks[n * P2 + l] = 1.f / (1.f + __expf(-s));
    }
  }
  __syncthreads();
  if (tid < 4) {
    float s = 0.f;
    for (int l = 0; l < P2; ++l) s += masks[tid * P2 + l];
    sn[tid] = s;
  }
  {
    const int k = tid >> 2, n = tid & 3;
    float s = 0.f;
    for (int l = 0; l < P2; ++l) s += x2s[k * P2 + l] * masks[n * P2 + l];
    Gm[k * 4 + n] = s;
  }
  __syncthreads();
  {
    const int c = tid >> 2, n = tid & 3;
    float s = 0.f;
    for (int k = 0; k < OC2; ++k) s += ctw[c * OC2 + k] * Gm[k * 4 + n];
    x3s[c * 4 + n] = s + ctb[c] * sn[n];
  }
  __syncthreads();
  {
    const int f = tid >> 2, m = tid & 3;
    float s = 0.f;
#pragma unroll
    for (int n = 0; n < 4; ++n) s += adjw[m * 4 + n] * x3s[f * 4 + n];
    Gm[f * 4 + m] = (s >= 0.f) ? s : 0.2f * s;
  }
  __syncthreads();
  {
    const int o = tid >> 2, n = tid & 3;
    float s = swb[o];
    for (int f = 0; f < OC2; ++f) s += sww[o * OC2 + f] * Gm[f * 4 + n];
    s = (s >= 0.f) ? s : 0.2f * s;
    x4s[o * 4 + n] = s + x3s[o * 4 + n];
  }
  __syncthreads();
  if (tid < OC2) {
    const int c = tid;
    float r0 = x4s[c*4+0], r1 = x4s[c*4+1], r2 = x4s[c*4+2], r3 = x4s[c*4+3];
    float mn = 3.0e38f;
    for (int d = 0; d < OC2; ++d) {
      float e2 = r0*x4s[d*4+0] + r1*x4s[d*4+1] + r2*x4s[d*4+2] + r3*x4s[d*4+3];
      mn = fminf(mn, e2);
    }
    float Z = 0.f, o0 = 0.f, o1 = 0.f, o2 = 0.f, o3 = 0.f;
    for (int d = 0; d < OC2; ++d) {
      float e2 = r0*x4s[d*4+0] + r1*x4s[d*4+1] + r2*x4s[d*4+2] + r3*x4s[d*4+3];
      float w = __expf(mn - e2);
      Z += w;
      o0 += w * x4s[d*4+0]; o1 += w * x4s[d*4+1];
      o2 += w * x4s[d*4+2]; o3 += w * x4s[d*4+3];
    }
    float iz = 1.f / Z;
    outs[c*4+0] = o0*iz; outs[c*4+1] = o1*iz;
    outs[c*4+2] = o2*iz; outs[c*4+3] = o3*iz;
  }
  __syncthreads();
  if (tid < 16) {
    const int m = tid >> 2, n = tid & 3;
    const float gamma = dgam[0];
    float s = dcob[m];
    for (int k = 0; k < OC2; ++k) {
      float xg = gamma * outs[k * 4 + n] + x4s[k * 4 + n];
      s += dcow[m * 128 + k] * xg;
      s += dcow[m * 128 + 64 + k] * x4s[k * 4 + n];
    }
    dadj[m * 4 + n] = 1.f / (1.f + __expf(-s));
  }
  __syncthreads();
  {
    const int c = tid >> 2, m = tid & 3;
    float s = 0.f;
#pragma unroll
    for (int n = 0; n < 4; ++n) s += x4s[c * 4 + n] * dadj[n * 4 + m];
    y1s[c * 4 + m] = (s >= 0.f) ? s : 0.2f * s;
  }
  __syncthreads();
  {
    const int o = tid >> 2, n = tid & 3;
    float s = dwb[o];
    for (int f = 0; f < OC2; ++f) s += dww[o * OC2 + f] * y1s[f * 4 + n];
    s = (s >= 0.f) ? s : 0.2f * s;
    x6out[(size_t)b * 256 + o * 4 + n] = s + x4s[o * 4 + n];
  }
}

extern "C" void kernel_launch(void* const* d_in, const int* in_sizes, int n_in,
                              void* d_out, int out_size, void* d_ws, size_t ws_size,
                              hipStream_t stream) {
  const float* x       = (const float*)d_in[0];
  const float* conv1_w = (const float*)d_in[1];
  const float* conv1_b = (const float*)d_in[2];
  const float* bn1_g   = (const float*)d_in[3];
  const float* bn1_b   = (const float*)d_in[4];
  const float* bn1_m   = (const float*)d_in[5];
  const float* bn1_v   = (const float*)d_in[6];
  const float* conv2_w = (const float*)d_in[7];
  const float* conv2_b = (const float*)d_in[8];
  const float* bn2_g   = (const float*)d_in[9];
  const float* bn2_b   = (const float*)d_in[10];
  const float* bn2_m   = (const float*)d_in[11];
  const float* bn2_v   = (const float*)d_in[12];
  const float* fc_sam  = (const float*)d_in[13];
  const float* ctw     = (const float*)d_in[14];
  const float* ctb     = (const float*)d_in[15];
  const float* adjw    = (const float*)d_in[16];
  const float* sww     = (const float*)d_in[17];
  const float* swb     = (const float*)d_in[18];
  const float* dcow    = (const float*)d_in[19];
  const float* dcob    = (const float*)d_in[20];
  const float* dgam    = (const float*)d_in[21];
  const float* dww     = (const float*)d_in[22];
  const float* dwb     = (const float*)d_in[23];

  const size_t n1f = (size_t)NT1 * KB1 * 32 * 16;   // 29696
  const size_t n2f = (size_t)NT2 * KB2 * 32 * 16;   // 18432
  char* wsb = (char*)d_ws;
  __bf16* w1f  = (__bf16*)wsb;
  __bf16* w2f  = (__bf16*)(wsb + n1f * 2);
  __bf16* x1bf = (__bf16*)(wsb + (n1f + n2f) * 2);

  float* x6out = (float*)d_out;
  float* x2out = (float*)d_out + X6SZ;

  const int prepN = (int)(n1f + n2f);
  prep_weights<<<(prepN + 255) / 256, 256, 0, stream>>>(conv1_w, conv2_w, w1f, w2f);

  conv1_kernel<<<NB, 256, 0, stream>>>(x, w1f, conv1_b,
                                       bn1_g, bn1_b, bn1_m, bn1_v, x1bf);

  conv2_graph_kernel<<<NB, 256, 0, stream>>>(x1bf, w2f, conv2_b,
                                             bn2_g, bn2_b, bn2_m, bn2_v,
                                             fc_sam, ctw, ctb, adjw, sww, swb,
                                             dcow, dcob, dgam, dww, dwb,
                                             x6out, x2out);
}
